// SelfAttention_64905545777627
// MI455X (gfx1250) — compile-verified, matrix operand order checked by reading, not run on hardware
//
#include <hip/hip_runtime.h>

// ---------------------------------------------------------------------------
// Self-attention (bs=4, seq=2048, embed=768, heads=12, d=64) for gfx1250.
// bf16 WMMA everywhere (v_wmma_f32_16x16x32_bf16), f32 accumulation.
//   1) cast x / W_qkv / W_proj to bf16
//   2) QKV GEMM (WMMA, 128x64 tile, BK=64) -> Q,K,V bf16 [b,h,s,d] (Q *1/8)
//   3) flash attention (WMMA QK^T + online softmax + WMMA PV)
//   4) proj GEMM (WMMA) -> f32 out + bias
// WMMA inner loops ordered so consecutive WMMAs hit different accumulators
// (avoids the 16-bit WMMA->WMMA RAW hazard NOPs). Next-stage global tiles are
// prefetched with global_prefetch_b8.
// ---------------------------------------------------------------------------

typedef __attribute__((ext_vector_type(16))) __bf16 v16bf;
typedef __attribute__((ext_vector_type(8)))  __bf16 v8bf;
typedef __attribute__((ext_vector_type(8)))  float  v8f;

#define BS    4
#define SEQ   2048
#define EMBED 768
#define NHEAD 12
#define HD    64
#define NBH   (BS * NHEAD)
#define MTOT  (BS * SEQ)          // 8192 rows
#define N_QKV (3 * EMBED)         // 2304

static __device__ __forceinline__ __bf16 f2bf(float f) {
  unsigned u = __builtin_bit_cast(unsigned, f);
  unsigned r = u + 0x7FFFu + ((u >> 16) & 1u);   // round-to-nearest-even
  unsigned short h = (unsigned short)(r >> 16);
  return __builtin_bit_cast(__bf16, h);
}

static __device__ __forceinline__ v16bf cat8(v8bf lo, v8bf hi) {
  return __builtin_shufflevector(lo, hi, 0,1,2,3,4,5,6,7,8,9,10,11,12,13,14,15);
}

static __device__ __forceinline__ v8f vzero8() {
  v8f v;
#pragma unroll
  for (int i = 0; i < 8; ++i) v[i] = 0.0f;
  return v;
}

static __device__ __forceinline__ v8f wmma_bf16(v16bf a, v16bf b, v8f c) {
  // (neg_a, A, neg_b, B, c_mod, C, reuse_a, reuse_b)
  return __builtin_amdgcn_wmma_f32_16x16x32_bf16(false, a, false, b,
                                                 (short)0, c, false, false);
}

// ---------------------------------------------------------------------------
__global__ void cast_f32_bf16(const float* __restrict__ s,
                              __bf16* __restrict__ d, int n) {
  int i = blockIdx.x * blockDim.x + threadIdx.x;
  int stride = gridDim.x * blockDim.x;
  for (; i < n; i += stride) d[i] = f2bf(s[i]);
}

// ---------------------------------------------------------------------------
// 128x64-tile bf16 GEMM, BK=64, 256 threads = 8 waves.
// Wave w owns rows [w*16, w*16+16) x all 64 cols: 4 accumulators, A-frag
// reused x4, consecutive WMMAs independent.
// MODE 0: QKV epilogue (bias, split into Q/K/V head-major, scale Q by 0.125)
// MODE 1: proj epilogue (bias, f32 out)
// ---------------------------------------------------------------------------
template <int MODE>
__global__ __launch_bounds__(256)
void gemm_bf16_128x64(const __bf16* __restrict__ A, const __bf16* __restrict__ B,
                      const float* __restrict__ bias, int M, int N, int K,
                      __bf16* __restrict__ outQ, __bf16* __restrict__ outK,
                      __bf16* __restrict__ outV, float* __restrict__ outF) {
  constexpr int LD = 72;                   // 144B rows -> 16B aligned b128 ops
  __shared__ __bf16 Alds[128 * LD];
  __shared__ __bf16 Blds[64 * LD];         // transposed: [n][k]

  const int tid  = threadIdx.x;
  const int lane = tid & 31;
  const int wave = tid >> 5;
  const int mr   = lane & 15;
  const int hlf  = lane >> 4;
  const int mBase = blockIdx.y * 128;
  const int nBase = blockIdx.x * 64;
  const int m0 = wave * 16;

  v8f acc[4] = {vzero8(), vzero8(), vzero8(), vzero8()};

  const int arow  = tid >> 1, acblk = (tid & 1) * 32;  // 128 rows x 2 chunks
  const int bk    = tid >> 2, bnblk = (tid & 3) * 16;  // 64 k    x 4 chunks

  for (int k0 = 0; k0 < K; k0 += 64) {
    __syncthreads();
    // A tile (row-major 128x64): 4 x b128 per thread
    const __bf16* ap = &A[(size_t)(mBase + arow) * K + k0 + acblk];
#pragma unroll
    for (int i = 0; i < 4; ++i)
      *(v8bf*)&Alds[arow * LD + acblk + i * 8] = *(const v8bf*)(ap + i * 8);
    // B tile transposed into LDS: Blds[n][k] = B[k0+k][nBase+n]
    const __bf16* bp = &B[(size_t)(k0 + bk) * N + nBase + bnblk];
    v8bf bw0 = *(const v8bf*)bp, bw1 = *(const v8bf*)(bp + 8);
#pragma unroll
    for (int i = 0; i < 8; ++i) {
      Blds[(bnblk + i) * LD + bk]     = bw0[i];
      Blds[(bnblk + 8 + i) * LD + bk] = bw1[i];
    }
    if (k0 + 64 < K) {                       // prefetch next stage
      __builtin_prefetch((const void*)(ap + 64), 0, 1);
      __builtin_prefetch((const void*)(bp + (size_t)64 * N), 0, 1);
    }
    __syncthreads();

#pragma unroll
    for (int kk = 0; kk < 2; ++kk) {         // k-step outer: indep. WMMAs
      v16bf a = cat8(*(const v8bf*)&Alds[(m0 + mr) * LD + kk * 32 + hlf * 8],
                     *(const v8bf*)&Alds[(m0 + mr) * LD + kk * 32 + 16 + hlf * 8]);
#pragma unroll
      for (int j = 0; j < 4; ++j) {
        const __bf16* bb = &Blds[(j * 16 + mr) * LD + kk * 32 + hlf * 8];
        acc[j] = wmma_bf16(a, cat8(*(const v8bf*)bb, *(const v8bf*)(bb + 16)),
                           acc[j]);
      }
    }
  }

#pragma unroll
  for (int j = 0; j < 4; ++j) {
    const int gn = nBase + j * 16 + mr;
    const float bv = bias[gn];
#pragma unroll
    for (int r = 0; r < 8; ++r) {
      const int gm = mBase + m0 + r + 8 * hlf;
      float v = acc[j][r] + bv;
      if constexpr (MODE == 0) {
        const int sel = gn / EMBED, c = gn % EMBED;
        const int head = c >> 6, d = c & 63;
        const int bi = gm >> 11, s = gm & 2047;
        const size_t off = (((size_t)bi * NHEAD + head) * SEQ + s) * HD + d;
        if (sel == 0)      outQ[off] = f2bf(v * 0.125f);  // fold 1/sqrt(64)
        else if (sel == 1) outK[off] = f2bf(v);
        else               outV[off] = f2bf(v);
      } else {
        outF[(size_t)gm * N + gn] = v;
      }
    }
  }
}

// ---------------------------------------------------------------------------
// Flash attention: block = (b,h) x 128 query rows, 8 waves x 16 rows each.
// 32 key tiles of 64; K tile row-major LDS == transposed-B layout for QK^T;
// V transposed in LDS for PV; P staged through a per-wave LDS slab fenced
// with s_wait_dscnt (wave-local round-trip, no block barrier needed).
// ---------------------------------------------------------------------------
__global__ __launch_bounds__(256)
void flash_attn(const __bf16* __restrict__ Q, const __bf16* __restrict__ Kx,
                const __bf16* __restrict__ V, const int* __restrict__ attmask,
                __bf16* __restrict__ Y) {
  constexpr int LDK = 72, LDV = 72, LDP = 72;   // 144B rows -> 16B aligned
  __shared__ __bf16 Klds[64 * LDK];
  __shared__ __bf16 Vlds[64 * LDV];             // transposed: [d][key]
  __shared__ __bf16 Plds[8 * 16 * LDP];         // per-wave 16x64 slab
  __shared__ int maskLds[64];

  const int bh = blockIdx.x;                    // 0..47
  const int b  = bh / NHEAD;
  const int h  = bh % NHEAD;
  const int qBase = blockIdx.y * 128;
  const int tid  = threadIdx.x;
  const int lane = tid & 31;
  const int wave = tid >> 5;
  const int mr   = lane & 15;
  const int hlf  = lane >> 4;

  // Q A-fragments for this wave's 16 rows (reused across all key tiles)
  const int qrow = qBase + wave * 16 + mr;
  const __bf16* qptr = Q + ((size_t)bh * SEQ + qrow) * HD;
  v16bf qfrag[2];
#pragma unroll
  for (int kk = 0; kk < 2; ++kk)
    qfrag[kk] = cat8(*(const v8bf*)(qptr + kk * 32 + hlf * 8),
                     *(const v8bf*)(qptr + kk * 32 + 16 + hlf * 8));

  v8f   ofrag[4] = {vzero8(), vzero8(), vzero8(), vzero8()};
  float mrun[8], lrun[8];
#pragma unroll
  for (int r = 0; r < 8; ++r) { mrun[r] = -3.0e38f; lrun[r] = 0.0f; }

  __bf16* pw = &Plds[wave * 16 * LDP];
  const int ckey = tid >> 2, cdblk = (tid & 3) * 16;

  for (int t = 0; t < SEQ / 64; ++t) {
    const int keyBase = t * 64;
    __syncthreads();
    // K tile row-major [key][d]
    const __bf16* kp = Kx + ((size_t)bh * SEQ + keyBase + ckey) * HD + cdblk;
    *(v8bf*)&Klds[ckey * LDK + cdblk]     = *(const v8bf*)kp;
    *(v8bf*)&Klds[ckey * LDK + cdblk + 8] = *(const v8bf*)(kp + 8);
    // V tile transposed [d][key]
    const __bf16* vp = V + ((size_t)bh * SEQ + keyBase + ckey) * HD + cdblk;
    v8bf v0 = *(const v8bf*)vp, v1 = *(const v8bf*)(vp + 8);
#pragma unroll
    for (int i = 0; i < 8; ++i) {
      Vlds[(cdblk + i) * LDV + ckey]     = v0[i];
      Vlds[(cdblk + 8 + i) * LDV + ckey] = v1[i];
    }
    if (tid < 64) maskLds[tid] = attmask[b * SEQ + keyBase + tid];
    if (t + 1 < SEQ / 64) {                    // prefetch next key/value tile
      __builtin_prefetch((const void*)(kp + 64 * HD), 0, 1);
      __builtin_prefetch((const void*)(vp + 64 * HD), 0, 1);
    }
    __syncthreads();

    // S = Q_tile @ K_tile^T: k-step outer so the 4 accumulators interleave
    v8f sfrag[4] = {vzero8(), vzero8(), vzero8(), vzero8()};
#pragma unroll
    for (int kk = 0; kk < 2; ++kk) {
#pragma unroll
      for (int n = 0; n < 4; ++n) {
        const __bf16* kb = &Klds[(n * 16 + mr) * LDK + kk * 32 + hlf * 8];
        sfrag[n] = wmma_bf16(qfrag[kk],
                             cat8(*(const v8bf*)kb, *(const v8bf*)(kb + 16)),
                             sfrag[n]);
      }
    }
    // mask: reference replaces masked-key scores with exactly -10000
#pragma unroll
    for (int n = 0; n < 4; ++n) {
      if (maskLds[n * 16 + mr] == 0) {
#pragma unroll
        for (int r = 0; r < 8; ++r) sfrag[n][r] = -10000.0f;
      }
    }

    // online softmax: row = r + 8*hlf; a row's 16 cols live in one half-wave
    float tmax[8];
#pragma unroll
    for (int r = 0; r < 8; ++r) {
      float m = sfrag[0][r];
      m = fmaxf(m, sfrag[1][r]); m = fmaxf(m, sfrag[2][r]); m = fmaxf(m, sfrag[3][r]);
      tmax[r] = m;
    }
#pragma unroll
    for (int xm = 1; xm < 16; xm <<= 1)
#pragma unroll
      for (int r = 0; r < 8; ++r)
        tmax[r] = fmaxf(tmax[r], __shfl_xor(tmax[r], xm, 32));

    float psum[8];
#pragma unroll
    for (int r = 0; r < 8; ++r) {
      const float mnew = fmaxf(mrun[r], tmax[r]);
      const float sc = __expf(mrun[r] - mnew);
      mrun[r] = mnew;
      lrun[r] *= sc;
#pragma unroll
      for (int f = 0; f < 4; ++f) ofrag[f][r] *= sc;
      psum[r] = 0.0f;
    }
#pragma unroll
    for (int n = 0; n < 4; ++n)
#pragma unroll
      for (int r = 0; r < 8; ++r) {
        const float p = __expf(sfrag[n][r] - mrun[r]);
        psum[r] += p;
        pw[(r + 8 * hlf) * LDP + n * 16 + mr] = f2bf(p);
      }
#pragma unroll
    for (int xm = 1; xm < 16; xm <<= 1)
#pragma unroll
      for (int r = 0; r < 8; ++r) psum[r] += __shfl_xor(psum[r], xm, 32);
#pragma unroll
    for (int r = 0; r < 8; ++r) lrun[r] += psum[r];

    // wave-local LDS round-trip for P (C-layout -> A-fragment layout)
    asm volatile("s_wait_dscnt 0" ::: "memory");

    // O += P @ V: k-step outer, A-frag reused x4, accumulators interleave
#pragma unroll
    for (int kk = 0; kk < 2; ++kk) {
      const __bf16* pa = &pw[mr * LDP + kk * 32 + hlf * 8];
      const v16bf af = cat8(*(const v8bf*)pa, *(const v8bf*)(pa + 16));
#pragma unroll
      for (int d0 = 0; d0 < 4; ++d0) {
        const __bf16* vb = &Vlds[(d0 * 16 + mr) * LDV + kk * 32 + hlf * 8];
        ofrag[d0] = wmma_bf16(af, cat8(*(const v8bf*)vb, *(const v8bf*)(vb + 16)),
                              ofrag[d0]);
      }
    }
  }

  // normalize and write y in [b, s, h*64+d] bf16 (A-matrix for proj GEMM)
#pragma unroll
  for (int r = 0; r < 8; ++r) {
    const float inv = 1.0f / lrun[r];
    const int gseq = qBase + wave * 16 + r + 8 * hlf;
#pragma unroll
    for (int d0 = 0; d0 < 4; ++d0) {
      Y[((size_t)b * SEQ + gseq) * EMBED + h * HD + d0 * 16 + mr] =
          f2bf(ofrag[d0][r] * inv);
    }
  }
}

// ---------------------------------------------------------------------------
extern "C" void kernel_launch(void* const* d_in, const int* in_sizes, int n_in,
                              void* d_out, int out_size, void* d_ws, size_t ws_size,
                              hipStream_t stream) {
  (void)in_sizes; (void)n_in; (void)out_size; (void)ws_size;
  const float* x      = (const float*)d_in[0];
  const int*   attm   = (const int*)d_in[1];
  const float* W_qkv  = (const float*)d_in[2];
  const float* b_qkv  = (const float*)d_in[3];
  const float* W_proj = (const float*)d_in[4];
  const float* b_proj = (const float*)d_in[5];
  float* out = (float*)d_out;

  // workspace layout (bf16 buffers, 256B-aligned offsets)
  char* ws = (char*)d_ws;
  const size_t SZ_X  = (size_t)MTOT * EMBED * 2;        // 12.6 MB
  const size_t SZ_WQ = (size_t)EMBED * N_QKV * 2;       //  3.5 MB
  const size_t SZ_WP = (size_t)EMBED * EMBED * 2;       //  1.2 MB
  const size_t SZ_H  = (size_t)NBH * SEQ * HD * 2;      // 12.6 MB each
  size_t off = 0;
  __bf16* xb    = (__bf16*)(ws + off); off += (SZ_X  + 255) & ~size_t(255);
  __bf16* wqkvb = (__bf16*)(ws + off); off += (SZ_WQ + 255) & ~size_t(255);
  __bf16* wprjb = (__bf16*)(ws + off); off += (SZ_WP + 255) & ~size_t(255);
  __bf16* qb    = (__bf16*)(ws + off); off += (SZ_H  + 255) & ~size_t(255);
  __bf16* kb    = (__bf16*)(ws + off); off += (SZ_H  + 255) & ~size_t(255);
  __bf16* vb    = (__bf16*)(ws + off); off += (SZ_H  + 255) & ~size_t(255);
  __bf16* yb    = (__bf16*)(ws + off);

  cast_f32_bf16<<<1024, 256, 0, stream>>>(x, xb, MTOT * EMBED);
  cast_f32_bf16<<<512, 256, 0, stream>>>(W_qkv, wqkvb, EMBED * N_QKV);
  cast_f32_bf16<<<256, 256, 0, stream>>>(W_proj, wprjb, EMBED * EMBED);

  // QKV projection: [8192 x 768] @ [768 x 2304]
  gemm_bf16_128x64<0><<<dim3(N_QKV / 64, MTOT / 128), 256, 0, stream>>>(
      xb, wqkvb, b_qkv, MTOT, N_QKV, EMBED, qb, kb, vb, nullptr);

  // attention: 48 (b,h) x 16 query chunks of 128
  flash_attn<<<dim3(NBH, SEQ / 128), 256, 0, stream>>>(qb, kb, vb, attm, yb);

  // output projection: [8192 x 768] @ [768 x 768] -> f32 out
  gemm_bf16_128x64<1><<<dim3(EMBED / 64, MTOT / 128), 256, 0, stream>>>(
      yb, wprjb, b_proj, MTOT, EMBED, EMBED, nullptr, nullptr, nullptr, out);
}
